// FaceDetector_50551765074113
// MI455X (gfx1250) — compile-verified, matrix operand order checked by reading, not run on hardware
//
#include <hip/hip_runtime.h>
#include <math.h>

// ---------------- types ----------------
typedef _Float16 h16;
typedef __attribute__((ext_vector_type(16))) _Float16 v16h;
typedef __attribute__((ext_vector_type(8)))  _Float16 v8h;
typedef __attribute__((ext_vector_type(8)))  float    v8f;

union Frag16 { v16h v; v8h h[2]; };

// ---------------- tiny prep kernels ----------------
// Fold BN into per-channel scale/shift:  y = conv*s + t ; s=g*rsqrt(v+eps), t=(b-m)*s+be
__global__ void prep_bn(const float* __restrict__ bi, const float* __restrict__ g,
                        const float* __restrict__ be, const float* __restrict__ m,
                        const float* __restrict__ v, float* __restrict__ s,
                        float* __restrict__ t, int C) {
    int c = blockIdx.x * blockDim.x + threadIdx.x;
    if (c < C) {
        float sc = g[c] * rsqrtf(v[c] + 1e-5f);
        s[c] = sc;
        t[c] = (bi[c] - m[c]) * sc + be[c];
    }
}

__global__ void prep_bias(const float* __restrict__ bi, float* __restrict__ s,
                          float* __restrict__ t, int C) {
    int c = blockIdx.x * blockDim.x + threadIdx.x;
    if (c < C) { s[c] = 1.0f; t[c] = bi[c]; }
}

// OIHW fp32 -> [O][ky][kx][I] f16  (contiguous K per output channel)
__global__ void prep_w3x3(const float* __restrict__ w, h16* __restrict__ o, int O, int I) {
    int i = blockIdx.x * 256 + threadIdx.x;
    int tot = O * I * 9;
    if (i >= tot) return;
    int oo = i / (I * 9);
    int r  = i % (I * 9);
    int ky = r / (3 * I);
    int r2 = r % (3 * I);
    int kx = r2 / I;
    int c  = r2 % I;
    o[i] = (h16)w[((oo * I + c) * 3 + ky) * 3 + kx];
}

__global__ void prep_w1x1(const float* __restrict__ w, h16* __restrict__ o, int n) {
    int i = blockIdx.x * 256 + threadIdx.x;
    if (i < n) o[i] = (h16)w[i];
}

// ---------------- conv1: 3->32, s2 p1, fused BN+ReLU, NCHW fp32 -> NHWC f16 ----------------
__global__ __launch_bounds__(256)
void conv1_direct(const float* __restrict__ x, const float* __restrict__ w,
                  const float* __restrict__ s, const float* __restrict__ t,
                  h16* __restrict__ out) {
    __shared__ float wsm[32 * 27];
    __shared__ float ssm[32], tsm[32];
    for (int i = threadIdx.x; i < 32 * 27; i += 256) wsm[i] = w[i];
    if (threadIdx.x < 32) { ssm[threadIdx.x] = s[threadIdx.x]; tsm[threadIdx.x] = t[threadIdx.x]; }
    __syncthreads();

    int idx = blockIdx.x * 256 + threadIdx.x;   // over 32*256*256 output pixels
    int b  = idx >> 16;
    int pp = idx & 65535;
    int oh = pp >> 8;
    int ow = pp & 255;

    float acc[32];
#pragma unroll
    for (int o = 0; o < 32; ++o) acc[o] = 0.f;

    for (int ky = 0; ky < 3; ++ky) {
        int ih = oh * 2 - 1 + ky;
        if ((unsigned)ih >= 512u) continue;
        for (int kx = 0; kx < 3; ++kx) {
            int iw = ow * 2 - 1 + kx;
            if ((unsigned)iw >= 512u) continue;
#pragma unroll
            for (int c = 0; c < 3; ++c) {
                float xv = x[((size_t)(b * 3 + c) * 512 + ih) * 512 + iw];
#pragma unroll
                for (int o = 0; o < 32; ++o)
                    acc[o] += xv * wsm[o * 27 + c * 9 + ky * 3 + kx];
            }
        }
    }
    h16* op = out + (size_t)idx * 32;
#pragma unroll
    for (int o = 0; o < 32; ++o) {
        float v = acc[o] * ssm[o] + tsm[o];
        op[o] = (h16)(v > 0.f ? v : 0.f);
    }
}

// ---------------- implicit-GEMM WMMA conv: 3x3 pad1, f16 NHWC in/out ----------------
// M = B*OH*OW (row-major b,oh,ow), N = COUT, K = 9*CIN in (ky,kx,c) order.
// Block tile 128x64; 8 wave32s as 4(M) x 2(N), each wave 32x32 via 2x2 wmma 16x16x32.
template <int CIN, int COUT, int OW, int IW, int STRIDE>
__global__ __launch_bounds__(256)
void conv_wmma(const h16* __restrict__ in, const h16* __restrict__ wk,
               const float* __restrict__ s, const float* __restrict__ t,
               h16* __restrict__ out) {
    constexpr int OHW = OW * OW;
    constexpr int KCH = CIN / 32;

    const int tid  = threadIdx.x;
    const int lane = tid & 31;
    const int wv   = tid >> 5;
    const int wm   = (wv >> 1) * 32;
    const int wn   = (wv & 1) * 32;
    const int m0   = blockIdx.x * 128;
    const int n0   = blockIdx.y * 64;

    __shared__ h16 As[128][40];   // 32 valid K-halves/row, padded to keep 16B alignment
    __shared__ h16 Bs[64][40];

    v8f zero = {0.f, 0.f, 0.f, 0.f, 0.f, 0.f, 0.f, 0.f};
    v8f acc[2][2];
    acc[0][0] = zero; acc[0][1] = zero; acc[1][0] = zero; acc[1][1] = zero;

    // per-row coords for the A-tile loader (threads 0..127)
    const int row = tid & 127;
    const int m   = m0 + row;
    const int b   = m / OHW;
    const int pp  = m % OHW;
    const int oh  = pp / OW;
    const int ow  = pp % OW;

    for (int tap = 0; tap < 9; ++tap) {
        const int ky = tap / 3, kx = tap % 3;
        for (int cb = 0; cb < KCH; ++cb) {
            // ---- stage A (activations) and B (weights) into LDS ----
            if (tid < 128) {
                const int ih = oh * STRIDE - 1 + ky;
                const int iw = ow * STRIDE - 1 + kx;
                uint4* dst = (uint4*)&As[tid][0];
                if ((unsigned)ih < (unsigned)IW && (unsigned)iw < (unsigned)IW) {
                    const uint4* src = (const uint4*)(in +
                        ((size_t)(b * IW + ih) * IW + iw) * CIN + cb * 32);
                    dst[0] = src[0]; dst[1] = src[1]; dst[2] = src[2]; dst[3] = src[3];
                } else {
                    uint4 z = {0u, 0u, 0u, 0u};
                    dst[0] = z; dst[1] = z; dst[2] = z; dst[3] = z;
                }
            } else if (tid < 192) {
                const int n = n0 + (tid - 128);
                const h16* wsrc = wk + (size_t)n * (9 * CIN) + tap * CIN + cb * 32;
                const uint4* src = (const uint4*)wsrc;
                uint4* dst = (uint4*)&Bs[tid - 128][0];
                dst[0] = src[0]; dst[1] = src[1]; dst[2] = src[2]; dst[3] = src[3];
                __builtin_prefetch(wsrc + 32, 0, 1);   // global_prefetch next K-chunk
            }
            __syncthreads();

            // ---- fragment loads per ISA 16-bit A 16x32 layout ----
            const int mr = lane & 15;
            const int kb = (lane < 16) ? 0 : 8;   // lanes 0-15: K 0-7/16-23 ; 16-31: K 8-15/24-31
            Frag16 a[2], bf[2];
#pragma unroll
            for (int i = 0; i < 2; ++i) {
                const int ma = wm + i * 16 + mr;
                a[i].h[0] = *(const v8h*)&As[ma][kb];
                a[i].h[1] = *(const v8h*)&As[ma][kb + 16];
                const int na = wn + i * 16 + mr;
                bf[i].h[0] = *(const v8h*)&Bs[na][kb];
                bf[i].h[1] = *(const v8h*)&Bs[na][kb + 16];
            }
#pragma unroll
            for (int i = 0; i < 2; ++i)
#pragma unroll
                for (int j = 0; j < 2; ++j)
                    acc[i][j] = __builtin_amdgcn_wmma_f32_16x16x32_f16(
                        false, a[i].v, false, bf[j].v, (short)0, acc[i][j], false, false);
            __syncthreads();
        }
    }

    // ---- epilogue: scale+shift+ReLU, f16 NHWC store (C/D layout: VGPR r -> M=r / 8+r) ----
#pragma unroll
    for (int i = 0; i < 2; ++i) {
        const int mbase = m0 + wm + i * 16 + ((lane < 16) ? 0 : 8);
#pragma unroll
        for (int j = 0; j < 2; ++j) {
            const int n = n0 + wn + j * 16 + (lane & 15);
            const float sc = s[n], tt = t[n];
#pragma unroll
            for (int r = 0; r < 8; ++r) {
                float v = acc[i][j][r] * sc + tt;
                v = v > 0.f ? v : 0.f;
                out[(size_t)(mbase + r) * COUT + n] = (h16)v;
            }
        }
    }
}

// ---------------- 1x1 head kernels (N=1 sigmoid, N=4 raw) ----------------
template <int NOUT, bool SIG>
__global__ __launch_bounds__(256)
void head2(const h16* __restrict__ act, const h16* __restrict__ wh,
           const float* __restrict__ bias, float* __restrict__ out) {
    __shared__ h16 wl[NOUT * 256];
    for (int i = threadIdx.x; i < NOUT * 256; i += 256) wl[i] = wh[i];
    __syncthreads();

    const size_t idx = (size_t)blockIdx.x * 256 + threadIdx.x;   // B*4096 pixels
    const h16* a = act + idx * 256;

    float acc[NOUT];
#pragma unroll
    for (int j = 0; j < NOUT; ++j) acc[j] = 0.f;

    for (int k = 0; k < 256; ++k) {
        float av = (float)a[k];
#pragma unroll
        for (int j = 0; j < NOUT; ++j) acc[j] += av * (float)wl[j * 256 + k];
    }
    if (SIG) {
        float z = acc[0] + bias[0];
        out[idx] = 1.f / (1.f + __expf(-z));
    } else {
#pragma unroll
        for (int j = 0; j < NOUT; ++j) out[idx * NOUT + j] = acc[j] + bias[j];
    }
}

// ---------------- per-image top-8 + gather + sigmoid*512 + threshold ----------------
__global__ __launch_bounds__(256)
void topk_kernel(const float* __restrict__ scores, const float* __restrict__ boxes,
                 float* __restrict__ d_out) {
    const int b = blockIdx.x;
    __shared__ float sc[4096];
    __shared__ float rv[256];
    __shared__ int   ri[256];
    for (int i = threadIdx.x; i < 4096; i += 256) sc[i] = scores[(size_t)b * 4096 + i];
    __syncthreads();

    for (int iter = 0; iter < 8; ++iter) {
        float best = -INFINITY;
        int   bi   = 0;
        for (int i = threadIdx.x; i < 4096; i += 256)
            if (sc[i] > best) { best = sc[i]; bi = i; }
        rv[threadIdx.x] = best;
        ri[threadIdx.x] = bi;
        __syncthreads();
        for (int off = 128; off > 0; off >>= 1) {
            if (threadIdx.x < off) {
                if (rv[threadIdx.x + off] > rv[threadIdx.x]) {
                    rv[threadIdx.x] = rv[threadIdx.x + off];
                    ri[threadIdx.x] = ri[threadIdx.x + off];
                }
            }
            __syncthreads();
        }
        if (threadIdx.x == 0) {
            const int   idx = ri[0];
            const float s   = rv[0];
            for (int j = 0; j < 4; ++j) {
                float bx = boxes[((size_t)b * 4096 + idx) * 4 + j];
                float v  = (s > 0.5f) ? (1.f / (1.f + expf(-bx))) * 512.f : 0.f;
                d_out[(b * 8 + iter) * 4 + j] = v;
            }
            sc[idx] = -INFINITY;
        }
        __syncthreads();
    }
}

// ---------------- launch ----------------
extern "C" void kernel_launch(void* const* d_in, const int* in_sizes, int n_in,
                              void* d_out, int out_size, void* d_ws, size_t ws_size,
                              hipStream_t stream) {
    (void)in_sizes; (void)n_in; (void)out_size; (void)ws_size;

    const float* x   = (const float*)d_in[0];
    const float* w1  = (const float*)d_in[1];
    const float* b1  = (const float*)d_in[2];
    const float* g1  = (const float*)d_in[3];
    const float* be1 = (const float*)d_in[4];
    const float* m1  = (const float*)d_in[5];
    const float* v1  = (const float*)d_in[6];
    const float* w2  = (const float*)d_in[7];
    const float* b2  = (const float*)d_in[8];
    const float* g2  = (const float*)d_in[9];
    const float* be2 = (const float*)d_in[10];
    const float* m2  = (const float*)d_in[11];
    const float* v2  = (const float*)d_in[12];
    const float* w3  = (const float*)d_in[13];
    const float* b3  = (const float*)d_in[14];
    const float* g3  = (const float*)d_in[15];
    const float* be3 = (const float*)d_in[16];
    const float* m3  = (const float*)d_in[17];
    const float* v3  = (const float*)d_in[18];
    const float* dw1 = (const float*)d_in[19];
    const float* db1 = (const float*)d_in[20];
    const float* dw2 = (const float*)d_in[21];
    const float* db2 = (const float*)d_in[22];
    const float* bw1 = (const float*)d_in[23];
    const float* bb1 = (const float*)d_in[24];
    const float* bw2 = (const float*)d_in[25];
    const float* bb2 = (const float*)d_in[26];

    // ---- workspace carve (256B aligned) ----
    char* p = (char*)d_ws;
    auto carve = [&](size_t bytes) -> void* {
        void* r = (void*)p;
        p += (bytes + 255) & ~(size_t)255;
        return r;
    };
    float* s1 = (float*)carve(32 * 4);   float* t1 = (float*)carve(32 * 4);
    float* s2 = (float*)carve(64 * 4);   float* t2 = (float*)carve(64 * 4);
    float* s3 = (float*)carve(128 * 4);  float* t3 = (float*)carve(128 * 4);
    float* sd = (float*)carve(256 * 4);  float* td = (float*)carve(256 * 4);
    float* sb = (float*)carve(256 * 4);  float* tb = (float*)carve(256 * 4);
    h16* wh2  = (h16*)carve((size_t)64 * 288 * 2);
    h16* wh3  = (h16*)carve((size_t)128 * 576 * 2);
    h16* whd1 = (h16*)carve((size_t)256 * 1152 * 2);
    h16* whb1 = (h16*)carve((size_t)256 * 1152 * 2);
    h16* whd2 = (h16*)carve((size_t)256 * 2);
    h16* whb2 = (h16*)carve((size_t)1024 * 2);
    float* scores = (float*)carve((size_t)32 * 4096 * 4);
    float* boxes  = (float*)carve((size_t)32 * 4096 * 4 * 4);
    h16* bufA = (h16*)carve((size_t)32 * 256 * 256 * 32 * 2);  // A1; reused as head mid D1
    h16* A2   = (h16*)carve((size_t)32 * 128 * 128 * 64 * 2);
    h16* A3   = (h16*)carve((size_t)32 * 64 * 64 * 128 * 2);
    h16* D1   = bufA;   // 67 MB needed; A1 (128 MB) is dead by then

    // ---- weight / BN prep ----
    prep_bn  <<<1, 64,  0, stream>>>(b1, g1, be1, m1, v1, s1, t1, 32);
    prep_bn  <<<1, 64,  0, stream>>>(b2, g2, be2, m2, v2, s2, t2, 64);
    prep_bn  <<<1, 128, 0, stream>>>(b3, g3, be3, m3, v3, s3, t3, 128);
    prep_bias<<<1, 256, 0, stream>>>(db1, sd, td, 256);
    prep_bias<<<1, 256, 0, stream>>>(bb1, sb, tb, 256);
    prep_w3x3<<<(64 * 32 * 9 + 255) / 256, 256, 0, stream>>>(w2, wh2, 64, 32);
    prep_w3x3<<<(128 * 64 * 9 + 255) / 256, 256, 0, stream>>>(w3, wh3, 128, 64);
    prep_w3x3<<<(256 * 128 * 9 + 255) / 256, 256, 0, stream>>>(dw1, whd1, 256, 128);
    prep_w3x3<<<(256 * 128 * 9 + 255) / 256, 256, 0, stream>>>(bw1, whb1, 256, 128);
    prep_w1x1<<<1, 256, 0, stream>>>(dw2, whd2, 256);
    prep_w1x1<<<4, 256, 0, stream>>>(bw2, whb2, 1024);

    // ---- backbone ----
    conv1_direct<<<(32 * 256 * 256) / 256, 256, 0, stream>>>(x, w1, s1, t1, bufA);
    // conv2: M=32*128*128=524288, N=64
    conv_wmma<32, 64, 128, 256, 2><<<dim3(4096, 1), 256, 0, stream>>>(bufA, wh2, s2, t2, A2);
    // conv3: M=32*64*64=131072, N=128
    conv_wmma<64, 128, 64, 128, 2><<<dim3(1024, 2), 256, 0, stream>>>(A2, wh3, s3, t3, A3);

    // ---- detection head ----
    conv_wmma<128, 256, 64, 64, 1><<<dim3(1024, 4), 256, 0, stream>>>(A3, whd1, sd, td, D1);
    head2<1, true><<<512, 256, 0, stream>>>(D1, whd2, db2, scores);

    // ---- bbox head (reuses D1 buffer) ----
    conv_wmma<128, 256, 64, 64, 1><<<dim3(1024, 4), 256, 0, stream>>>(A3, whb1, sb, tb, D1);
    head2<4, false><<<512, 256, 0, stream>>>(D1, whb2, bb2, boxes);

    // ---- top-8 + gather + sigmoid*512 + threshold -> [32,8,4] ----
    topk_kernel<<<32, 256, 0, stream>>>(scores, boxes, (float*)d_out);
}
